// HyperNodeGCN_38087769981667
// MI455X (gfx1250) — matrix operand order repaired and checked by reading, not compiled, not run on hardware
//
#include <hip/hip_runtime.h>
#include <math.h>
#include <stdint.h>

// ---- problem constants (from the reference) ----
#define BS   8
#define LSEQ 107
#define NNF  17
#define NEF  9
#define NH   128
#define EH   32
#define H3   32
#define NLAY 10
#define PC   32
#define TLAY 3
#define THD  8
#define FFD  256
#define NCLS 3
#define DD   160          // NH + PC
#define HD   20           // DD / THD
#define NN   (BS*LSEQ)    // 856
#define EE   2896         // total edges
#define NSEQ 1696         // sequence edges
#define EPSV 1e-5f
#define KSPLIT 4          // K-dimension split of the edge-message GEMM
#define JSEG  (EH / KSPLIT)   // 8 j-values per segment

typedef __attribute__((ext_vector_type(16))) _Float16 v16h;
typedef __attribute__((ext_vector_type(8)))  float    v8f;

static __device__ __forceinline__ _Float16 f2h(float f) { return (_Float16)f; }
static __device__ __forceinline__ float gelu_exact(float x) {
  return 0.5f * x * (1.0f + erff(x * 0.70710678118654752f));
}

// ================= ChebConv helpers (tiny, VALU) =================
__global__ void k_deg(const int* __restrict__ ss, float* deg, int ne) {
  int e = blockIdx.x * blockDim.x + threadIdx.x;
  if (e < ne) atomicAdd(&deg[ss[e]], 1.0f);
}
__global__ void k_dinv(const float* __restrict__ deg, float* dinv, int n) {
  int i = blockIdx.x * blockDim.x + threadIdx.x;
  if (i < n) dinv[i] = (deg[i] > 0.0f) ? rsqrtf(fmaxf(deg[i], 1e-12f)) : 0.0f;
}
__global__ void k_lap(const int* __restrict__ ss, const int* __restrict__ sd,
                      const float* __restrict__ dinv, const float* __restrict__ in,
                      float* out, int ne, int f) {
  int e = blockIdx.x * blockDim.x + threadIdx.x;
  if (e >= ne) return;
  int s = ss[e], d = sd[e];
  float w = -(dinv[s] * dinv[d]);
  for (int j = 0; j < f; ++j) atomicAdd(&out[d * f + j], w * in[s * f + j]);
}
__global__ void k_tx2fix(const float* __restrict__ x, float* tx2, int n) {
  int i = blockIdx.x * blockDim.x + threadIdx.x;
  if (i < n) tx2[i] = 2.0f * tx2[i] - x[i];
}
__global__ void k_cheb(const float* __restrict__ x, const float* __restrict__ tx1,
                       const float* __restrict__ tx2, const float* __restrict__ w,
                       const float* __restrict__ b, float* h) {
  int i = blockIdx.x * blockDim.x + threadIdx.x;
  if (i >= NN * NH) return;
  int n = i / NH, o = i % NH;
  const float* w0 = w;
  const float* w1 = w + NNF * NH;
  const float* w2 = w + 2 * NNF * NH;
  float acc = b[o];
  for (int j = 0; j < NNF; ++j)
    acc += x[n*NNF+j]*w0[j*NH+o] + tx1[n*NNF+j]*w1[j*NH+o] + tx2[n*NNF+j]*w2[j*NH+o];
  h[i] = acc;
}

// ================= small dense helpers =================
__global__ void k_edge_enc(const float* __restrict__ ein, const float* __restrict__ w,
                           const float* __restrict__ b, float* ea) {
  int i = blockIdx.x * blockDim.x + threadIdx.x;
  if (i >= EE * EH) return;
  int e = i / EH, o = i % EH;
  float acc = b[o];
  for (int j = 0; j < NEF; ++j) acc += ein[e*NEF+j] * w[j*EH+o];
  ea[i] = acc;
}
__global__ void k_edge_a(const float* __restrict__ ea, const float* __restrict__ w1,
                         const float* __restrict__ b1, float* a) {
  int i = blockIdx.x * blockDim.x + threadIdx.x;
  if (i >= EE * H3) return;
  int e = i / H3, o = i % H3;
  float acc = b1[o];
  for (int j = 0; j < EH; ++j) acc += ea[e*EH+j] * w1[j*H3+o];
  a[i] = acc;
}
__global__ void k_ln_gelu(const float* __restrict__ h, const float* __restrict__ g,
                          const float* __restrict__ b, float* out) {
  int n = blockIdx.x * blockDim.x + threadIdx.x;
  if (n >= NN) return;
  float m = 0.f;
  for (int j = 0; j < NH; ++j) m += h[n*NH+j];
  m *= (1.0f / NH);
  float v = 0.f;
  for (int j = 0; j < NH; ++j) { float d = h[n*NH+j] - m; v += d * d; }
  v *= (1.0f / NH);
  float inv = rsqrtf(v + EPSV);
  for (int j = 0; j < NH; ++j)
    out[n*NH+j] = gelu_exact((h[n*NH+j] - m) * inv * g[j] + b[j]);
}
__global__ void k_combine(float* __restrict__ h, const float* __restrict__ acc, int addflag) {
  int i = blockIdx.x * blockDim.x + threadIdx.x;
  if (i >= NN * NH) return;
  h[i] = addflag ? (h[i] + acc[i]) : acc[i];
}
__global__ void k_msg_bias(const float* __restrict__ hin, const float* __restrict__ b2,
                           const int* __restrict__ src, const int* __restrict__ dst,
                           float* accum) {
  int i = blockIdx.x * blockDim.x + threadIdx.x;
  if (i >= EE * NH) return;
  int e = i / NH, o = i % NH;
  int s = src[e];
  float acc = 0.f;
  for (int j = 0; j < NH; ++j) acc += hin[s*NH+j] * b2[j*NH+o];
  atomicAdd(&accum[dst[e]*NH + o], acc);
}

// ================= WMMA kernels =================
// Heavy NNConv message GEMM: D[e][o] = sum_k A[e][k]*B[k][o], K = 4096, where
//   A[e][k] = a[e][k>>7] * hin[src[e]][k&127]  (built in registers: hin row packed
//   to f16 once into F[0..3], per chunk one pk_mul by a_e[j])
//   B[k][o] = w2[(k>>7)*16384 + (k&127)*128 + o] (async-DMA'd to LDS, double-buffered)
// Grid: (E/16, 128/16, KSPLIT); each block reduces an 8-j segment, atomic scatter out.
__global__ void __launch_bounds__(32)
k_msg_wmma(const float* __restrict__ a, const float* __restrict__ hin,
           const float* __restrict__ w2, const int* __restrict__ src,
           const int* __restrict__ dst, float* accum) {
  __shared__ float lbuf[2][32 * 16];          // two 2KB B-tile buffers
  const int te  = blockIdx.x;                 // 16 edges (E/16 == 181 exact)
  const int to  = blockIdx.y;                 // 16 output cols
  const int j0  = blockIdx.z * JSEG;          // K segment
  const int lane = threadIdx.x;               // wave32
  const int m15 = lane & 15;
  const int hiA = (lane >> 4) ? 8 : 0;        // ISA 16-bit A layout half-split
  const int hiB = (lane >> 4) ? 16 : 0;       // unused; B gather uses hiA (same k order as A)
  const int ecur = te * 16 + m15;
  const int sA = src[ecur];

  // ---- preload this lane's source-node row, packed to f16 A-fragments ----
  v16h F[4];
#pragma unroll
  for (int q = 0; q < 4; ++q) {
    const float* r0 = hin + (size_t)sA * NH + q * 32 + hiA;
    float4 a0 = *(const float4*)(r0);
    float4 a1 = *(const float4*)(r0 + 4);
    float4 b0 = *(const float4*)(r0 + 16);
    float4 b1 = *(const float4*)(r0 + 20);
    F[q][0]=f2h(a0.x); F[q][1]=f2h(a0.y); F[q][2]=f2h(a0.z); F[q][3]=f2h(a0.w);
    F[q][4]=f2h(a1.x); F[q][5]=f2h(a1.y); F[q][6]=f2h(a1.z); F[q][7]=f2h(a1.w);
    F[q][8]=f2h(b0.x); F[q][9]=f2h(b0.y); F[q][10]=f2h(b0.z); F[q][11]=f2h(b0.w);
    F[q][12]=f2h(b1.x); F[q][13]=f2h(b1.y); F[q][14]=f2h(b1.z); F[q][15]=f2h(b1.w);
  }

  const unsigned lb0 = (unsigned)(uintptr_t)&lbuf[0][0];
  const unsigned lb1 = (unsigned)(uintptr_t)&lbuf[1][0];
  const int br = lane >> 2;                   // row-group within async instruction
  const int bs4 = (lane & 3) * 4;             // 16-byte segment

  // async-DMA one 2KB B tile (rows ibase..ibase+31, cols to*16..+15) into `buf`
  auto issueB = [&](int j, int ibase, int buf) {
    unsigned lbase = buf ? lb1 : lb0;
#pragma unroll
    for (int u = 0; u < 4; ++u) {
      int r = u * 8 + br;
      const float* gp = w2 + (size_t)j * (NH * NH) + (size_t)(ibase + r) * NH
                           + to * 16 + bs4;
      unsigned lds = lbase + (unsigned)(r * 64 + bs4 * 4);
      unsigned long long ga = (unsigned long long)(uintptr_t)gp;
      asm volatile("global_load_async_to_lds_b128 %0, %1, off"
                   :: "v"(lds), "v"(ga) : "memory");
    }
  };

  issueB(j0, 0, 0);
  v8f c = {};
  for (int jj = 0; jj < JSEG; ++jj) {
    const int j = j0 + jj;
    float aj = a[(size_t)ecur * H3 + j];
    _Float16 ajh = f2h(aj);
    v16h ajv;
#pragma unroll
    for (int v = 0; v < 16; ++v) ajv[v] = ajh;
#pragma unroll
    for (int q = 0; q < 4; ++q) {
      const int cc = jj * 4 + q;
      // prefetch next tile into the other buffer
      if (q < 3)                issueB(j, (q + 1) * 32, (cc + 1) & 1);
      else if (jj < JSEG - 1)   issueB(j + 1, 0, (cc + 1) & 1);
      // async loads retire in order: <=4 frees current tile, leaves next in flight
      if (cc < JSEG * 4 - 1) asm volatile("s_wait_asynccnt 0x4" ::: "memory");
      else                   asm volatile("s_wait_asynccnt 0x0" ::: "memory");

      const float* lb = lbuf[cc & 1];
      // gather B fragment: distinct temps so all ds_loads can issue before one wait
      float tb[16];
#pragma unroll
      for (int v = 0; v < 16; ++v) {
        int g = v >> 1, hh = v & 1;
        int kk = (v & 7) + ((v >> 3) << 4) + hiA;
        tb[v] = lb[kk * 16 + m15];
      }
      v16h bf;
#pragma unroll
      for (int v = 0; v < 16; ++v) bf[v] = f2h(tb[v]);
      v16h af = ajv * F[q];                  // v_pk_mul_f16
      c = __builtin_amdgcn_wmma_f32_16x16x32_f16(false, af, false, bf,
                                                 (short)0, c, false, false);
    }
  }
  // C/D layout: vgpr g -> row g (lanes 0-15) / g+8 (lanes 16-31); col = lane&15
#pragma unroll
  for (int g = 0; g < 8; ++g) {
    int er = te * 16 + g + ((lane >> 4) ? 8 : 0);
    atomicAdd(&accum[(size_t)dst[er] * NH + (to * 16 + m15)], c[g]);
  }
}

// Generic f32 GEMM via f16 WMMA, f32 accumulate: C[M,N] = A[M,K]@B[K,N] + bias, opt ReLU
// Requires K % 32 == 0 (true for all call sites: 128/160/256).
__global__ void __launch_bounds__(32)
k_gemm_wmma(const float* __restrict__ A, const float* __restrict__ B,
            const float* __restrict__ bias, float* __restrict__ C,
            int M, int N, int K, int relu) {
  const int tm = blockIdx.x, tn = blockIdx.y;
  const int lane = threadIdx.x;
  const int m15 = lane & 15;
  const int hiA = (lane >> 4) ? 8 : 0;
  const int hiB = (lane >> 4) ? 16 : 0;
  const int row = tm * 16 + m15;
  const int col = tn * 16 + m15;
  const bool rok = row < M;
  const bool cok = col < N;
  v8f c = {};
  for (int k0 = 0; k0 < K; k0 += 32) {
    v16h af, bf;
    float4 a0, a1, a2, a3;
    if (rok) {
      const float* ap = A + (size_t)row * K + k0 + hiA;
      a0 = *(const float4*)(ap);
      a1 = *(const float4*)(ap + 4);
      a2 = *(const float4*)(ap + 16);
      a3 = *(const float4*)(ap + 20);
    } else {
      a0 = a1 = a2 = a3 = make_float4(0.f, 0.f, 0.f, 0.f);
    }
    af[0]=f2h(a0.x); af[1]=f2h(a0.y); af[2]=f2h(a0.z); af[3]=f2h(a0.w);
    af[4]=f2h(a1.x); af[5]=f2h(a1.y); af[6]=f2h(a1.z); af[7]=f2h(a1.w);
    af[8]=f2h(a2.x); af[9]=f2h(a2.y); af[10]=f2h(a2.z); af[11]=f2h(a2.w);
    af[12]=f2h(a3.x); af[13]=f2h(a3.y); af[14]=f2h(a3.z); af[15]=f2h(a3.w);
    if (k0 + 32 < K) __builtin_prefetch(B + (size_t)(k0 + 32) * N + col, 0, 1);
    float tb[16];
#pragma unroll
    for (int v = 0; v < 16; ++v) {
      int g = v >> 1, hh = v & 1;
      int k = k0 + (v & 7) + ((v >> 3) << 4) + hiA;
      tb[v] = cok ? B[(size_t)k * N + col] : 0.f;
    }
#pragma unroll
    for (int v = 0; v < 16; ++v) bf[v] = f2h(tb[v]);
    c = __builtin_amdgcn_wmma_f32_16x16x32_f16(false, af, false, bf,
                                               (short)0, c, false, false);
  }
#pragma unroll
  for (int g = 0; g < 8; ++g) {
    int r = tm * 16 + g + ((lane >> 4) ? 8 : 0);
    if (r < M && cok) {
      float v = c[g] + (bias ? bias[col] : 0.f);
      if (relu) v = fmaxf(v, 0.f);
      C[(size_t)r * N + col] = v;
    }
  }
}

// ================= transformer-side VALU kernels =================
__global__ void k_build_t(const float* __restrict__ h, float* t) {
  int i = blockIdx.x * blockDim.x + threadIdx.x;
  if (i >= NN * DD) return;
  int n = i / DD, d = i % DD;
  float v;
  if (d < NH) v = h[n * NH + d];
  else {
    int li = n % LSEQ;
    int cpos = d - NH;
    int p = cpos >> 1;
    float div = __expf((2.0f * p) * (-logf(10000.0f) / (float)PC));
    float ang = (float)li * div;
    v = (cpos & 1) ? cosf(ang) : sinf(ang);
  }
  t[i] = v;
}
__global__ void k_attn(const float* __restrict__ qkv, float* __restrict__ out) {
  int idx = blockIdx.x * blockDim.x + threadIdx.x;
  if (idx >= BS * THD * LSEQ) return;
  int b = idx / (THD * LSEQ);
  int rem = idx % (THD * LSEQ);
  int hh = rem / LSEQ;
  int q = rem % LSEQ;
  int nq = b * LSEQ + q;
  const float scale = rsqrtf((float)HD);
  float qv[HD];
  for (int d = 0; d < HD; ++d) qv[d] = qkv[(size_t)nq * (3*DD) + hh * HD + d];
  float s[LSEQ];
  float mx = -1e30f;
  for (int kk = 0; kk < LSEQ; ++kk) {
    const float* kvp = qkv + (size_t)(b * LSEQ + kk) * (3*DD) + DD + hh * HD;
    float dot = 0.f;
    for (int d = 0; d < HD; ++d) dot += qv[d] * kvp[d];
    dot *= scale;
    s[kk] = dot;
    mx = fmaxf(mx, dot);
  }
  float se = 0.f;
  for (int kk = 0; kk < LSEQ; ++kk) { s[kk] = __expf(s[kk] - mx); se += s[kk]; }
  float inv = 1.0f / se;
  for (int d = 0; d < HD; ++d) {
    float acc = 0.f;
    for (int kk = 0; kk < LSEQ; ++kk)
      acc += s[kk] * qkv[(size_t)(b * LSEQ + kk) * (3*DD) + 2*DD + hh * HD + d];
    out[(size_t)nq * DD + hh * HD + d] = acc * inv;
  }
}
__global__ void k_addln(float* __restrict__ t, const float* __restrict__ o,
                        const float* __restrict__ g, const float* __restrict__ b) {
  int n = blockIdx.x * blockDim.x + threadIdx.x;
  if (n >= NN) return;
  float m = 0.f;
  for (int j = 0; j < DD; ++j) m += t[n*DD+j] + o[n*DD+j];
  m *= (1.0f / DD);
  float v = 0.f;
  for (int j = 0; j < DD; ++j) { float z = t[n*DD+j] + o[n*DD+j] - m; v += z*z; }
  v *= (1.0f / DD);
  float inv = rsqrtf(v + EPSV);
  for (int j = 0; j < DD; ++j) {
    float z = t[n*DD+j] + o[n*DD+j];
    t[n*DD+j] = (z - m) * inv * g[j] + b[j];
  }
}

// ================= host orchestration =================
extern "C" void kernel_launch(void* const* d_in, const int* in_sizes, int n_in,
                              void* d_out, int out_size, void* d_ws, size_t ws_size,
                              hipStream_t stream) {
  const float* x         = (const float*)d_in[0];
  const float* edge_attr = (const float*)d_in[1];
  const float* cheb_w    = (const float*)d_in[2];
  const float* cheb_b    = (const float*)d_in[3];
  const float* ee_w      = (const float*)d_in[4];
  const float* ee_b      = (const float*)d_in[5];
  const float* mlp_w1    = (const float*)d_in[6];
  const float* mlp_b1    = (const float*)d_in[7];
  const float* mlp_w2    = (const float*)d_in[8];
  const float* mlp_b2    = (const float*)d_in[9];
  const float* root_w    = (const float*)d_in[10];
  const float* conv_b    = (const float*)d_in[11];
  const float* ln_g      = (const float*)d_in[12];
  const float* ln_b      = (const float*)d_in[13];
  const float* ain_w     = (const float*)d_in[14];
  const float* ain_b     = (const float*)d_in[15];
  const float* aout_w    = (const float*)d_in[16];
  const float* aout_b    = (const float*)d_in[17];
  const float* ff_w1     = (const float*)d_in[18];
  const float* ff_b1     = (const float*)d_in[19];
  const float* ff_w2     = (const float*)d_in[20];
  const float* ff_b2     = (const float*)d_in[21];
  const float* tln1_g    = (const float*)d_in[22];
  const float* tln1_b    = (const float*)d_in[23];
  const float* tln2_g    = (const float*)d_in[24];
  const float* tln2_b    = (const float*)d_in[25];
  const float* lin_w     = (const float*)d_in[26];
  const float* lin_b     = (const float*)d_in[27];
  const int*   seq_ei    = (const int*)d_in[28];
  const int*   ei        = (const int*)d_in[29];
  const int* ss  = seq_ei;
  const int* sd  = seq_ei + NSEQ;
  const int* src = ei;
  const int* dst = ei + EE;

  float* ws = (float*)d_ws;
  size_t p = 0;
  float* deg  = ws + p; p += NN;
  float* dinv = ws + p; p += NN;
  float* tx1  = ws + p; p += (size_t)NN * NNF;
  float* tx2  = ws + p; p += (size_t)NN * NNF;
  float* h    = ws + p; p += (size_t)NN * NH;
  float* gbuf = ws + p; p += (size_t)NN * NH;
  float* acc  = ws + p; p += (size_t)NN * NH;
  float* ea   = ws + p; p += (size_t)EE * EH;
  float* abuf = ws + p; p += (size_t)EE * H3;
  float* t    = ws + p; p += (size_t)NN * DD;
  float* qkv  = ws + p; p += (size_t)NN * 3 * DD;
  float* ao   = ws + p; p += (size_t)NN * DD;
  float* proj = ws + p; p += (size_t)NN * DD;
  float* ffh  = ws + p; p += (size_t)NN * FFD;

  // ---- ChebConv ----
  hipMemsetAsync(deg, 0, NN * sizeof(float), stream);
  hipMemsetAsync(tx1, 0, (size_t)NN * NNF * sizeof(float), stream);
  hipMemsetAsync(tx2, 0, (size_t)NN * NNF * sizeof(float), stream);
  k_deg<<<(NSEQ + 255) / 256, 256, 0, stream>>>(ss, deg, NSEQ);
  k_dinv<<<(NN + 255) / 256, 256, 0, stream>>>(deg, dinv, NN);
  k_lap<<<(NSEQ + 255) / 256, 256, 0, stream>>>(ss, sd, dinv, x, tx1, NSEQ, NNF);
  k_lap<<<(NSEQ + 255) / 256, 256, 0, stream>>>(ss, sd, dinv, tx1, tx2, NSEQ, NNF);
  k_tx2fix<<<(NN * NNF + 255) / 256, 256, 0, stream>>>(x, tx2, NN * NNF);
  k_cheb<<<(NN * NH + 255) / 256, 256, 0, stream>>>(x, tx1, tx2, cheb_w, cheb_b, h);

  // ---- edge encoder ----
  k_edge_enc<<<(EE * EH + 255) / 256, 256, 0, stream>>>(edge_attr, ee_w, ee_b, ea);

  // ---- NNConv stack (WMMA + async-LDS heavy path) ----
  for (int l = 0; l < NLAY; ++l) {
    const float* hin = h;
    if (l > 0) {
      k_ln_gelu<<<(NN + 255) / 256, 256, 0, stream>>>(h, ln_g + l * NH, ln_b + l * NH, gbuf);
      hin = gbuf;
    }
    k_edge_a<<<(EE * H3 + 255) / 256, 256, 0, stream>>>(
        ea, mlp_w1 + (size_t)l * EH * H3, mlp_b1 + (size_t)l * H3, abuf);
    dim3 gr((NN + 15) / 16, NH / 16);
    k_gemm_wmma<<<gr, 32, 0, stream>>>(hin, root_w + (size_t)l * NH * NH,
                                       conv_b + (size_t)l * NH, acc, NN, NH, NH, 0);
    dim3 gm(EE / 16, NH / 16, KSPLIT);
    k_msg_wmma<<<gm, 32, 0, stream>>>(abuf, hin,
                                      mlp_w2 + (size_t)l * EH * NH * NH, src, dst, acc);
    k_msg_bias<<<(EE * NH + 255) / 256, 256, 0, stream>>>(
        hin, mlp_b2 + (size_t)l * NH * NH, src, dst, acc);
    k_combine<<<(NN * NH + 255) / 256, 256, 0, stream>>>(h, acc, l > 0 ? 1 : 0);
  }
  k_ln_gelu<<<(NN + 255) / 256, 256, 0, stream>>>(h, ln_g, ln_b, gbuf);
  k_build_t<<<(NN * DD + 255) / 256, 256, 0, stream>>>(gbuf, t);

  // ---- transformer encoder ----
  for (int l = 0; l < TLAY; ++l) {
    dim3 gq((NN + 15) / 16, (3 * DD) / 16);
    k_gemm_wmma<<<gq, 32, 0, stream>>>(t, ain_w + (size_t)l * DD * 3 * DD,
                                       ain_b + (size_t)l * 3 * DD, qkv, NN, 3 * DD, DD, 0);
    k_attn<<<(BS * THD * LSEQ + 63) / 64, 64, 0, stream>>>(qkv, ao);
    dim3 go((NN + 15) / 16, DD / 16);
    k_gemm_wmma<<<go, 32, 0, stream>>>(ao, aout_w + (size_t)l * DD * DD,
                                       aout_b + (size_t)l * DD, proj, NN, DD, DD, 0);
    k_addln<<<(NN + 255) / 256, 256, 0, stream>>>(t, proj, tln1_g + (size_t)l * DD,
                                                  tln1_b + (size_t)l * DD);
    dim3 gf1((NN + 15) / 16, FFD / 16);
    k_gemm_wmma<<<gf1, 32, 0, stream>>>(t, ff_w1 + (size_t)l * DD * FFD,
                                        ff_b1 + (size_t)l * FFD, ffh, NN, FFD, DD, 1);
    dim3 gf2((NN + 15) / 16, DD / 16);
    k_gemm_wmma<<<gf2, 32, 0, stream>>>(ffh, ff_w2 + (size_t)l * FFD * DD,
                                        ff_b2 + (size_t)l * DD, proj, NN, DD, FFD, 0);
    k_addln<<<(NN + 255) / 256, 256, 0, stream>>>(t, proj, tln2_g + (size_t)l * DD,
                                                  tln2_b + (size_t)l * DD);
  }

  // ---- classifier head ----
  dim3 gl((NN + 15) / 16, (NCLS + 15) / 16);
  k_gemm_wmma<<<gl, 32, 0, stream>>>(t, lin_w, lin_b, (float*)d_out, NN, NCLS, DD, 0);
}
